// SeNaTraStage_35107062678371
// MI455X (gfx1250) — compile-verified, matrix operand order checked by reading, not run on hardware
//
#include <hip/hip_runtime.h>
#include <hip/hip_bf16.h>
#include <math.h>

// ---------------- problem constants ----------------
#define BB   8
#define HI_  56
#define WI_  56
#define HO_  28
#define WO_  28
#define DIN  96
#define DOUT 192
#define NIN  (HI_*WI_)     // 3136
#define NOUTN (HO_*WO_)    // 784
#define HIDG (2*DOUT)      // 384
#define HIDB (3*DOUT)      // 576
#define NH   6
#define HD   32
#define NAW  7

typedef __attribute__((ext_vector_type(16))) _Float16 v16h;
typedef __attribute__((ext_vector_type(8)))  _Float16 v8h;
typedef __attribute__((ext_vector_type(8)))  float    v8f;

__device__ __forceinline__ float wave_sum(float v) {
    #pragma unroll
    for (int o = 16; o; o >>= 1) v += __shfl_xor(v, o, 32);
    return v;
}
__device__ __forceinline__ float wave_max(float v) {
    #pragma unroll
    for (int o = 16; o; o >>= 1) v = fmaxf(v, __shfl_xor(v, o, 32));
    return v;
}

// Build a 16-half WMMA fragment from an LDS row: halves [kb..kb+7] and [kb+16..kb+23].
__device__ __forceinline__ v16h frag16(const _Float16* rowp, int kb) {
    v8h lo = *(const v8h*)(rowp + kb);
    v8h hi = *(const v8h*)(rowp + kb + 16);
    return __builtin_shufflevector(lo, hi, 0,1,2,3,4,5,6,7,8,9,10,11,12,13,14,15);
}

// Load 16 consecutive-k f32 elements (vectorized when stride==1).
__device__ __forceinline__ void load16(const float* __restrict__ p, long long stride, float out[16]) {
    if (stride == 1) {
        const float4* p4 = (const float4*)p;
        float4 f0 = p4[0], f1 = p4[1], f2 = p4[2], f3 = p4[3];
        out[0]=f0.x; out[1]=f0.y; out[2]=f0.z; out[3]=f0.w;
        out[4]=f1.x; out[5]=f1.y; out[6]=f1.z; out[7]=f1.w;
        out[8]=f2.x; out[9]=f2.y; out[10]=f2.z; out[11]=f2.w;
        out[12]=f3.x; out[13]=f3.y; out[14]=f3.z; out[15]=f3.w;
    } else {
        #pragma unroll
        for (int j = 0; j < 16; ++j) out[j] = p[(long long)j * stride];
    }
}

// Convert 16 f32 -> f16 and store as two 16B LDS writes.
__device__ __forceinline__ void stage16(_Float16* dst, const float v[16]) {
    v8h h0, h1;
    #pragma unroll
    for (int j = 0; j < 8; ++j) { h0[j] = (_Float16)v[j]; h1[j] = (_Float16)v[j + 8]; }
    *(v8h*)(dst)     = h0;
    *(v8h*)(dst + 8) = h1;
}

// ---------------- generic strided batched WMMA GEMM ----------------
// C[b][m][n] = act( alpha * sum_k A(m,k)*B(k,n) + bias[n] ) + beta*Cin[b][m][n]
// A(m,k) = A[b*bsA + m*saM + k*saK], B(k,n) = B[b*bsB + k*sbK + n*sbN]
// 128 threads = 4 waves; 64x64 macro-tile; K multiple of 32; double-buffered LDS.
#define LDSROW 40   // halves per LDS row (32 data + 8 pad) -> 80B row stride

__global__ void k_gemm_wmma(const float* __restrict__ A, long long saM, long long saK, long long bsA,
                            const float* __restrict__ Bm, long long sbK, long long sbN, long long bsB,
                            const float* __restrict__ bias,
                            const float* __restrict__ Cin, float beta,
                            float* __restrict__ C, long long bsC,
                            int M, int N, int K,
                            const float* __restrict__ alpha_ptr, int do_exp, int act)
{
    __shared__ __align__(16) _Float16 As[2][64 * LDSROW];
    __shared__ __align__(16) _Float16 Bs[2][64 * LDSROW];

    const int bz = blockIdx.z;
    A  += (long long)bz * bsA;
    Bm += (long long)bz * bsB;
    C  += (long long)bz * bsC;
    const float* Cr = Cin ? (Cin + (long long)bz * bsC) : nullptr;

    const int m0 = blockIdx.x * 64, n0 = blockIdx.y * 64;
    const int tid  = threadIdx.x;
    const int lane = tid & 31;
    const int wave = tid >> 5;
    const int l16  = lane & 15;
    const int half = lane >> 4;
    const int wm = wave >> 1;        // row quadrant
    const int wn = wave & 1;         // col quadrant

    // cooperative-load assignment: row = tid/2 (0..63), k-segment = tid&1 (16 k's each)
    const int ra  = tid >> 1;
    const int seg = tid & 1;
    const int mg = min(m0 + ra, M - 1);          // clamped global row for A
    const int ng = min(n0 + ra, N - 1);          // clamped global col for B
    const float* apg = A  + (long long)mg * saM;
    const float* bpg = Bm + (long long)ng * sbN;
    _Float16* stA0 = &As[0][ra * LDSROW + seg * 16];
    _Float16* stB0 = &Bs[0][ra * LDSROW + seg * 16];
    _Float16* stA1 = &As[1][ra * LDSROW + seg * 16];
    _Float16* stB1 = &Bs[1][ra * LDSROW + seg * 16];

    v8f acc00 = {0,0,0,0,0,0,0,0}, acc01 = {0,0,0,0,0,0,0,0};
    v8f acc10 = {0,0,0,0,0,0,0,0}, acc11 = {0,0,0,0,0,0,0,0};
    const int kb = half * 8;

    // prologue: stage chunk 0 into buffer 0
    float at[16], bt[16];
    load16(apg + (long long)(seg * 16) * saK, saK, at);
    load16(bpg + (long long)(seg * 16) * sbK, sbK, bt);
    stage16(stA0, at);
    stage16(stB0, bt);
    __syncthreads();

    int cur = 0;
    for (int k0 = 0; k0 < K; k0 += 32) {
        const bool has_next = (k0 + 32) < K;
        if (has_next) {
            const long long kN = k0 + 32 + seg * 16;
            load16(apg + kN * saK, saK, at);   // overlaps with WMMAs below
            load16(bpg + kN * sbK, sbK, bt);
            __builtin_prefetch(apg + (kN + 32) * saK);
            __builtin_prefetch(bpg + (kN + 32) * sbK);
        }

        const _Float16* aB = As[cur];
        const _Float16* bB = Bs[cur];
        v16h a0 = frag16(aB + (wm * 32 + l16)      * LDSROW, kb);
        v16h a1 = frag16(aB + (wm * 32 + 16 + l16) * LDSROW, kb);
        v16h b0 = frag16(bB + (wn * 32 + l16)      * LDSROW, kb);
        v16h b1 = frag16(bB + (wn * 32 + 16 + l16) * LDSROW, kb);
        acc00 = __builtin_amdgcn_wmma_f32_16x16x32_f16(false, a0, false, b0, (short)0, acc00, false, false);
        acc01 = __builtin_amdgcn_wmma_f32_16x16x32_f16(false, a0, false, b1, (short)0, acc01, false, false);
        acc10 = __builtin_amdgcn_wmma_f32_16x16x32_f16(false, a1, false, b0, (short)0, acc10, false, false);
        acc11 = __builtin_amdgcn_wmma_f32_16x16x32_f16(false, a1, false, b1, (short)0, acc11, false, false);

        if (has_next) {
            // stage next chunk into the other buffer; single barrier per iteration:
            // it both publishes these stores and guarantees all waves finished
            // reading buf[cur] before it gets overwritten in the next iteration.
            if (cur) { stage16(stA0, at); stage16(stB0, bt); }
            else     { stage16(stA1, at); stage16(stB1, bt); }
            __syncthreads();
            cur ^= 1;
        }
    }

    float alpha = 1.f;
    if (alpha_ptr) { alpha = *alpha_ptr; if (do_exp) alpha = expf(alpha); }

    #pragma unroll
    for (int ti = 0; ti < 2; ++ti) {
        #pragma unroll
        for (int tj = 0; tj < 2; ++tj) {
            const v8f acc = (ti == 0) ? (tj == 0 ? acc00 : acc01)
                                      : (tj == 0 ? acc10 : acc11);
            const int col = n0 + wn * 32 + tj * 16 + l16;
            const int rowb = m0 + wm * 32 + ti * 16;
            if (col < N) {
                const float bc = bias ? bias[col] : 0.f;
                if (rowb + 15 < M) {
                    #pragma unroll
                    for (int r = 0; r < 8; ++r) {
                        const long long ci = (long long)(rowb + r + half * 8) * N + col;
                        float v = acc[r] * alpha + bc;
                        if (act == 1) v = 0.5f * v * (1.f + erff(v * 0.70710678118f));
                        if (Cr) v += beta * Cr[ci];
                        C[ci] = v;
                    }
                } else {
                    #pragma unroll
                    for (int r = 0; r < 8; ++r) {
                        const int row = rowb + r + half * 8;
                        if (row < M) {
                            const long long ci = (long long)row * N + col;
                            float v = acc[r] * alpha + bc;
                            if (act == 1) v = 0.5f * v * (1.f + erff(v * 0.70710678118f));
                            if (Cr) v += beta * Cr[ci];
                            C[ci] = v;
                        }
                    }
                }
            }
        }
    }
}

// ---------------- seed conv 3x3 stride2 pad1, NHWC->(b,n,co) ----------------
__global__ void k_conv_seed(const float* __restrict__ x, const float* __restrict__ w,
                            float* __restrict__ out)
{
    int t = blockIdx.x * blockDim.x + threadIdx.x;
    if (t >= BB * NOUTN * DOUT) return;
    int co = t % DOUT;
    int n  = (t / DOUT) % NOUTN;
    int b  = t / (DOUT * NOUTN);
    int ho = n / WO_, wo = n % WO_;
    float s = 0.f;
    for (int ci = 0; ci < DIN; ++ci) {
        const float* wp = w + ((long long)co * DIN + ci) * 9;
        #pragma unroll
        for (int kh = 0; kh < 3; ++kh) {
            int r = 2 * ho - 1 + kh;
            if (r < 0 || r >= HI_) continue;
            #pragma unroll
            for (int kw = 0; kw < 3; ++kw) {
                int c = 2 * wo - 1 + kw;
                if (c < 0 || c >= WI_) continue;
                s += x[((long long)b * NIN + r * WI_ + c) * DIN + ci] * wp[kh * 3 + kw];
            }
        }
    }
    out[t] = s;
}

// ---------------- LayerNorm (wave per row), optional residual-add ----------------
__global__ void k_ln(const float* __restrict__ in, float* __restrict__ out,
                     const float* __restrict__ g, const float* __restrict__ bt,
                     int rows, int D, int residual)
{
    int lane = threadIdx.x & 31;
    int row  = blockIdx.x * (blockDim.x >> 5) + (threadIdx.x >> 5);
    if (row >= rows) return;
    const float* p = in + (long long)row * D;
    const int cnt = D >> 5;
    float xs[6];
    float s = 0.f;
    for (int i = 0; i < cnt; ++i) { xs[i] = p[lane + 32 * i]; s += xs[i]; }
    s = wave_sum(s);
    const float mean = s / (float)D;
    float v = 0.f;
    for (int i = 0; i < cnt; ++i) { float d = xs[i] - mean; v += d * d; }
    v = wave_sum(v);
    const float inv = rsqrtf(v / (float)D + 1e-5f);
    float* q = out + (long long)row * D;
    for (int i = 0; i < cnt; ++i) {
        int dc = lane + 32 * i;
        float val = (xs[i] - mean) * inv * g[dc] + bt[dc];
        if (residual) q[dc] += val; else q[dc] = val;
    }
}

// ---------------- grouped softmax with on-the-fly QIDX/M_LOC ----------------
__global__ void k_gsoftmax(float* __restrict__ att, const float* __restrict__ rpb)
{
    int lane = threadIdx.x & 31;
    int row  = blockIdx.x * (blockDim.x >> 5) + (threadIdx.x >> 5);
    if (row >= BB * NIN) return;
    int n = row % NIN;
    int r = n / WI_, c = n % WI_;
    int r0 = r >> 1, c0 = c >> 1;
    int ois[9];
    #pragma unroll
    for (int kk = 0; kk < 9; ++kk) {
        int dr = kk / 3 - 1, dc = kk % 3 - 1;
        int rr = min(max(r0 + dr, 0), HO_ - 1);
        int cc = min(max(c0 + dc, 0), WO_ - 1);
        ois[kk] = rr * WO_ + cc;
    }
    float* p = att + (long long)row * NOUTN;
    float mx = -1e30f;
    for (int m = lane; m < NOUTN; m += 32) {
        float bias = -10000.f;
        #pragma unroll
        for (int t = 0; t < 9; ++t) if (ois[t] == m) bias = rpb[t];
        mx = fmaxf(mx, p[m] + bias);
    }
    mx = wave_max(mx);
    float sum = 0.f;
    for (int m = lane; m < NOUTN; m += 32) {
        float bias = -10000.f;
        #pragma unroll
        for (int t = 0; t < 9; ++t) if (ois[t] == m) bias = rpb[t];
        float e = expf(p[m] + bias - mx);
        p[m] = e; sum += e;
    }
    sum = wave_sum(sum);
    const float inv = 1.f / sum;
    for (int m = lane; m < NOUTN; m += 32) p[m] *= inv;
}

// ---------------- column sums of attention (LDS tree reduce) ----------------
__global__ void k_colsum(const float* __restrict__ att, float* __restrict__ cs)
{
    __shared__ float red[128];
    int bm = blockIdx.x;
    int b = bm / NOUTN, m = bm % NOUTN;
    const float* p = att + (long long)b * NIN * NOUTN + m;
    float s = 0.f;
    for (int n = threadIdx.x; n < NIN; n += 128) s += p[(long long)n * NOUTN];
    red[threadIdx.x] = s;
    __syncthreads();
    for (int o = 64; o > 0; o >>= 1) {
        if ((int)threadIdx.x < o) red[threadIdx.x] += red[threadIdx.x + o];
        __syncthreads();
    }
    if (threadIdx.x == 0) cs[bm] = red[0];
}

__global__ void k_adown(float* __restrict__ att, const float* __restrict__ cs)
{
    long long t = (long long)blockIdx.x * blockDim.x + threadIdx.x;
    if (t >= (long long)BB * NIN * NOUTN) return;
    int m = (int)(t % NOUTN);
    int b = (int)(t / ((long long)NIN * NOUTN));
    att[t] /= (cs[b * NOUTN + m] + 1e-8f);
}

// ---------------- qkv split + RoPE -> [b][h][n][d] layouts ----------------
__global__ void k_rope(const float* __restrict__ qkv, float* __restrict__ qr,
                       float* __restrict__ kr, float* __restrict__ vr)
{
    int t = blockIdx.x * blockDim.x + threadIdx.x;
    if (t >= BB * NOUTN * NH * (HD / 2)) return;
    int d2 = t % (HD / 2);
    int h  = (t / (HD / 2)) % NH;
    int n  = (t / ((HD / 2) * NH)) % NOUTN;
    int b  =  t / ((HD / 2) * NH * NOUTN);
    float inv = expf(-(2.f * (float)d2 / (float)HD) * logf(10000.f));
    float ang = (float)n * inv;
    float co = cosf(ang), si = sinf(ang);
    long long base = ((long long)b * NOUTN + n) * (3 * DOUT);
    long long ob   = (((long long)b * NH + h) * NOUTN + n) * HD + 2 * d2;
    float x0 = qkv[base + 0 * DOUT + h * HD + 2 * d2];
    float x1 = qkv[base + 0 * DOUT + h * HD + 2 * d2 + 1];
    qr[ob]     = x0 * co - x1 * si;
    qr[ob + 1] = x1 * co + x0 * si;
    x0 = qkv[base + 1 * DOUT + h * HD + 2 * d2];
    x1 = qkv[base + 1 * DOUT + h * HD + 2 * d2 + 1];
    kr[ob]     = x0 * co - x1 * si;
    kr[ob + 1] = x1 * co + x0 * si;
    vr[ob]     = qkv[base + 2 * DOUT + h * HD + 2 * d2];
    vr[ob + 1] = qkv[base + 2 * DOUT + h * HD + 2 * d2 + 1];
}

// ---------------- 7x7 neighborhood attention, wave per (b,h,n), lane=channel ----------------
__global__ void k_na_attn(const float* __restrict__ qr, const float* __restrict__ kr,
                          const float* __restrict__ vr, float* __restrict__ o)
{
    int lane = threadIdx.x & 31;
    int row  = blockIdx.x * (blockDim.x >> 5) + (threadIdx.x >> 5);
    if (row >= BB * NH * NOUTN) return;
    int n = row % NOUTN;
    int h = (row / NOUTN) % NH;
    int b =  row / (NOUTN * NH);
    int i = n / WO_, j = n % WO_;
    int ci = min(max(i, 3), HO_ - 4);
    int cj = min(max(j, 3), WO_ - 4);
    long long base = ((long long)b * NH + h) * NOUTN;
    float qd = qr[(base + n) * HD + lane];
    float m_run = -1e30f, l_run = 0.f, acc = 0.f;
    for (int kk = 0; kk < NAW * NAW; ++kk) {
        int rr = ci + kk / NAW - 3;
        int cc = cj + kk % NAW - 3;
        int n2 = rr * WO_ + cc;
        float t = wave_sum(qd * kr[(base + n2) * HD + lane]);
        float s = t * 0.17677669529f;
        float m_new = fmaxf(m_run, s);
        float scl = expf(m_run - m_new);
        float e   = expf(s - m_new);
        l_run = l_run * scl + e;
        acc   = acc   * scl + e * vr[(base + n2) * HD + lane];
        m_run = m_new;
    }
    o[((long long)b * NOUTN + n) * DOUT + h * HD + lane] = acc / l_run;
}

// ---------------- workspace layout (floats) ----------------
static constexpr size_t OFF_XN  = 0;
static constexpr size_t OFF_K   = OFF_XN  + (size_t)BB * NIN  * DIN;
static constexpr size_t OFF_V   = OFF_K   + (size_t)BB * NIN  * DIN;
static constexpr size_t OFF_Q   = OFF_V   + (size_t)BB * NIN  * DOUT;
static constexpr size_t OFF_ATT = OFF_Q   + (size_t)BB * NOUTN * DIN;
static constexpr size_t OFF_CS  = OFF_ATT + (size_t)BB * NIN  * NOUTN;
static constexpr size_t OFF_TA  = OFF_CS  + (size_t)BB * NOUTN;
static constexpr size_t OFF_TB  = OFF_TA  + (size_t)BB * NOUTN * HIDB;
static constexpr size_t OFF_TC  = OFF_TB  + (size_t)BB * NOUTN * DOUT;
static constexpr size_t OFF_QR  = OFF_TC  + (size_t)BB * NOUTN * DOUT;
static constexpr size_t OFF_KR  = OFF_QR  + (size_t)BB * NOUTN * DOUT;
static constexpr size_t OFF_VR  = OFF_KR  + (size_t)BB * NOUTN * DOUT;

static inline void gemm(hipStream_t st,
                        const float* A, long long saM, long long saK, long long bsA,
                        const float* Bm, long long sbK, long long sbN, long long bsB,
                        const float* bias, const float* Cin, float beta,
                        float* C, long long bsC, int M, int N, int K,
                        const float* alpha_ptr, int do_exp, int act)
{
    dim3 grid((M + 63) / 64, (N + 63) / 64, BB);
    k_gemm_wmma<<<grid, 128, 0, st>>>(A, saM, saK, bsA, Bm, sbK, sbN, bsB,
                                      bias, Cin, beta, C, bsC, M, N, K,
                                      alpha_ptr, do_exp, act);
}

extern "C" void kernel_launch(void* const* d_in, const int* in_sizes, int n_in,
                              void* d_out, int out_size, void* d_ws, size_t ws_size,
                              hipStream_t stream)
{
    const float* x      = (const float*)d_in[0];
    const float* convw  = (const float*)d_in[1];
    const float* qw     = (const float*)d_in[2];   // (96,192)
    const float* kw     = (const float*)d_in[3];   // (96,96)
    const float* vw     = (const float*)d_in[4];   // (192,96)
    const float* lin_g  = (const float*)d_in[5];
    const float* lin_b  = (const float*)d_in[6];
    const float* lout_g = (const float*)d_in[7];
    const float* lout_b = (const float*)d_in[8];
    const float* lat_g  = (const float*)d_in[9];
    const float* lat_b  = (const float*)d_in[10];
    const float* lml_g  = (const float*)d_in[11];
    const float* lml_b  = (const float*)d_in[12];
    const float* mw1    = (const float*)d_in[13];  // (384,192)
    const float* mb1    = (const float*)d_in[14];
    const float* mw2    = (const float*)d_in[15];  // (192,384)
    const float* mb2    = (const float*)d_in[16];
    const float* tau    = (const float*)d_in[17];
    const float* rpb    = (const float*)d_in[18];
    const float* bl1g   = (const float*)d_in[19];
    const float* bl1b   = (const float*)d_in[20];
    const float* bl2g   = (const float*)d_in[21];
    const float* bl2b   = (const float*)d_in[22];
    const float* bqkvw  = (const float*)d_in[23];  // (2,576,192)
    const float* bprojw = (const float*)d_in[24];  // (2,192,192)
    const float* bprojb = (const float*)d_in[25];
    const float* bm1w   = (const float*)d_in[26];  // (2,576,192)
    const float* bm1b   = (const float*)d_in[27];
    const float* bm2w   = (const float*)d_in[28];  // (2,192,576)
    const float* bm2b   = (const float*)d_in[29];

    float* xo  = (float*)d_out;
    float* ws  = (float*)d_ws;
    float* xn  = ws + OFF_XN;
    float* kb  = ws + OFF_K;
    float* vb  = ws + OFF_V;
    float* qb  = ws + OFF_Q;
    float* att = ws + OFF_ATT;
    float* cs  = ws + OFF_CS;
    float* ta  = ws + OFF_TA;
    float* tb  = ws + OFF_TB;
    float* tc  = ws + OFF_TC;
    float* qr  = ws + OFF_QR;
    float* kr  = ws + OFF_KR;
    float* vr  = ws + OFF_VR;

    // 1) seed conv -> tb (b,n,c); x_out = LN(tb)
    k_conv_seed<<<(BB * NOUTN * DOUT + 255) / 256, 256, 0, stream>>>(x, convw, tb);
    k_ln<<<(BB * NOUTN) / 8, 256, 0, stream>>>(tb, xo, lout_g, lout_b, BB * NOUTN, DOUT, 0);

    // 2) xn = LN(x); k = xn @ kwT; v = xn @ vwT
    k_ln<<<(BB * NIN) / 8, 256, 0, stream>>>(x, xn, lin_g, lin_b, BB * NIN, DIN, 0);
    gemm(stream, xn, DIN, 1, (long long)NIN * DIN, kw, 1, DIN, 0,
         nullptr, nullptr, 0.f, kb, (long long)NIN * DIN, NIN, DIN, DIN, nullptr, 0, 0);
    gemm(stream, xn, DIN, 1, (long long)NIN * DIN, vw, 1, DIN, 0,
         nullptr, nullptr, 0.f, vb, (long long)NIN * DOUT, NIN, DOUT, DIN, nullptr, 0, 0);

    // 3) grouped attention iterations
    for (int it = 0; it < 3; ++it) {
        k_ln<<<(BB * NOUTN) / 8, 256, 0, stream>>>(xo, tc, lout_g, lout_b, BB * NOUTN, DOUT, 0);
        gemm(stream, tc, DOUT, 1, (long long)NOUTN * DOUT, qw, 1, DOUT, 0,
             nullptr, nullptr, 0.f, qb, (long long)NOUTN * DIN, NOUTN, DIN, DOUT, nullptr, 0, 0);
        gemm(stream, kb, DIN, 1, (long long)NIN * DIN, qb, 1, DIN, (long long)NOUTN * DIN,
             nullptr, nullptr, 0.f, att, (long long)NIN * NOUTN, NIN, NOUTN, DIN, tau, 1, 0);
        k_gsoftmax<<<(BB * NIN) / 8, 256, 0, stream>>>(att, rpb);
        k_colsum<<<BB * NOUTN, 128, 0, stream>>>(att, cs);
        {
            long long tot = (long long)BB * NIN * NOUTN;
            k_adown<<<(unsigned)((tot + 255) / 256), 256, 0, stream>>>(att, cs);
        }
        gemm(stream, att, 1, NOUTN, (long long)NIN * NOUTN, vb, DOUT, 1, (long long)NIN * DOUT,
             nullptr, nullptr, 0.f, tb, (long long)NOUTN * DOUT, NOUTN, DOUT, NIN, nullptr, 0, 0);
        k_ln<<<(BB * NOUTN) / 8, 256, 0, stream>>>(tb, xo, lat_g, lat_b, BB * NOUTN, DOUT, 1);
        gemm(stream, xo, DOUT, 1, (long long)NOUTN * DOUT, mw1, 1, DOUT, 0,
             mb1, nullptr, 0.f, ta, (long long)NOUTN * HIDG, NOUTN, HIDG, DOUT, nullptr, 0, 1);
        gemm(stream, ta, HIDG, 1, (long long)NOUTN * HIDG, mw2, 1, HIDG, 0,
             mb2, nullptr, 0.f, tb, (long long)NOUTN * DOUT, NOUTN, DOUT, HIDG, nullptr, 0, 0);
        k_ln<<<(BB * NOUTN) / 8, 256, 0, stream>>>(tb, xo, lml_g, lml_b, BB * NOUTN, DOUT, 1);
    }

    // 4) neighborhood-attention transformer blocks
    for (int d = 0; d < 2; ++d) {
        k_ln<<<(BB * NOUTN) / 8, 256, 0, stream>>>(xo, tc, bl1g + d * DOUT, bl1b + d * DOUT,
                                                   BB * NOUTN, DOUT, 0);
        gemm(stream, tc, DOUT, 1, (long long)NOUTN * DOUT,
             bqkvw + (long long)d * HIDB * DOUT, 1, DOUT, 0,
             nullptr, nullptr, 0.f, ta, (long long)NOUTN * HIDB, NOUTN, HIDB, DOUT, nullptr, 0, 0);
        k_rope<<<(BB * NOUTN * NH * (HD / 2)) / 256, 256, 0, stream>>>(ta, qr, kr, vr);
        k_na_attn<<<(BB * NH * NOUTN) / 8, 256, 0, stream>>>(qr, kr, vr, tb);
        gemm(stream, tb, DOUT, 1, (long long)NOUTN * DOUT,
             bprojw + (long long)d * DOUT * DOUT, 1, DOUT, 0,
             bprojb + d * DOUT, xo, 1.f, xo, (long long)NOUTN * DOUT,
             NOUTN, DOUT, DOUT, nullptr, 0, 0);
        k_ln<<<(BB * NOUTN) / 8, 256, 0, stream>>>(xo, tc, bl2g + d * DOUT, bl2b + d * DOUT,
                                                   BB * NOUTN, DOUT, 0);
        gemm(stream, tc, DOUT, 1, (long long)NOUTN * DOUT,
             bm1w + (long long)d * HIDB * DOUT, 1, DOUT, 0,
             bm1b + d * HIDB, nullptr, 0.f, ta, (long long)NOUTN * HIDB,
             NOUTN, HIDB, DOUT, nullptr, 0, 1);
        gemm(stream, ta, HIDB, 1, (long long)NOUTN * HIDB,
             bm2w + (long long)d * DOUT * HIDB, 1, HIDB, 0,
             bm2b + d * DOUT, xo, 1.f, xo, (long long)NOUTN * DOUT,
             NOUTN, DOUT, HIDB, nullptr, 0, 0);
    }
    (void)in_sizes; (void)n_in; (void)out_size; (void)ws_size;
}
